// BaseAttention_23321672417345
// MI455X (gfx1250) — matrix operand order repaired and checked by reading, not compiled, not run on hardware
//
#include <hip/hip_runtime.h>
#include <hip/hip_bf16.h>
#include <math.h>

// MI455X (gfx1250) wave32 WMMA attention.
// All matmuls run on v_wmma_f32_16x16x32_bf16 (fp32 accumulate).
// Problem fits in 192MB L2 -> compute bound. B operands are staged into LDS
// with global_load_async_to_lds_b128 (ASYNCcnt) and shared by 8 waves.
// Each wave computes a 32x64 tile (2 M-strips) so every LDS B fragment feeds
// two WMMAs and every A fragment feeds four.

typedef __attribute__((ext_vector_type(16))) __bf16 v16bf;
typedef __attribute__((ext_vector_type(8)))  __bf16 v8bf;
typedef __attribute__((ext_vector_type(8)))  float  v8f;

#define HID   1024
#define SEQ   2048
#define BATCH 4

#define BN  64   // B-tile rows (output columns per block)
#define BK  64   // k-chunk per stage
#define BKP 72   // padded LDS row stride (36 dwords -> conflict-free lane quads)
#define TM  2    // M-strips per wave (32 rows)
#define BM  (256 / 32 * 16 * TM)   // 256 rows per block

// Async copy: GVS mode -> vdst = LDS byte offset VGPR, vaddr = 32-bit byte
// offset VGPR, saddr = 64-bit SGPR base.  Tracked by ASYNCcnt.
#define ASYNC_B128(lds, voff, sbase)                                  \
  asm volatile("global_load_async_to_lds_b128 %0, %1, %2"             \
               :: "v"(lds), "v"(voff), "s"(sbase) : "memory")
#define WAIT_ASYNCCNT(n) asm volatile("s_wait_asynccnt " #n ::: "memory")

// ---------------------------------------------------------------------------
// Fragment loaders (per-lane layouts per CDNA5 ISA 7.12.2)
// ---------------------------------------------------------------------------
static __device__ __forceinline__ v16bf load_a_frag(const __bf16* __restrict__ base,
                                                    int ld, int row0, int k0, int lane) {
  int m = lane & 15, h = lane >> 4;
  const __bf16* r = base + (size_t)(row0 + m) * ld + k0;
  union { v16bf v; v8bf h8[2]; } u;
  u.h8[0] = *(const v8bf*)(r + 8 * h);
  u.h8[1] = *(const v8bf*)(r + 16 + 8 * h);
  return u.v;
}

static __device__ __forceinline__ v16bf load_b_frag_lds(const __bf16* tile,
                                                        int nbase, int kk, int lane) {
  int n = lane & 15, h = lane >> 4;
  const __bf16* r = tile + (nbase + n) * BKP + kk + 8 * h;
  union { v16bf v; v8bf h8[2]; } u;
  u.h8[0] = *(const v8bf*)(r);      // 16B aligned -> ds_load_b128
  u.h8[1] = *(const v8bf*)(r + 16);
  return u.v;
}

// ---------------------------------------------------------------------------
// Kernel 1: fp32 -> bf16 cast, 4 elements per thread
// ---------------------------------------------------------------------------
__global__ __launch_bounds__(256) void k_cast_bf16(const float* __restrict__ src,
                                                   __bf16* __restrict__ dst, int n) {
  int i = (blockIdx.x * blockDim.x + threadIdx.x) * 4;
  if (i + 3 < n) {
    float4 f = *(const float4*)(src + i);
    dst[i + 0] = (__bf16)f.x;
    dst[i + 1] = (__bf16)f.y;
    dst[i + 2] = (__bf16)f.z;
    dst[i + 3] = (__bf16)f.w;
  }
}

// ---------------------------------------------------------------------------
// Kernel 2: Q/K/V projection.  y = x @ W^T.  Block = 8 waves = 256 rows x 64
// cols; per wave 32x64.  B tile (64x64) double-buffered in LDS (async).
// which==2 stores V transposed: Vt[b][d][s].
// ---------------------------------------------------------------------------
__global__ __launch_bounds__(256) void k_qkv_gemm(const __bf16* __restrict__ xb,
                                                  const __bf16* __restrict__ Wb,
                                                  __bf16* __restrict__ Qb,
                                                  __bf16* __restrict__ Kb,
                                                  __bf16* __restrict__ Vt) {
  __shared__ __bf16 btile[2][BN * BKP];
  const int tid  = threadIdx.x;
  const int lane = tid & 31;
  const int wave = tid >> 5;
  const int n0   = blockIdx.x * BN;
  const int row0 = blockIdx.y * BM + wave * 16 * TM;
  const int which = blockIdx.z;
  const __bf16* W = Wb + (size_t)which * HID * HID;

  const int f0 = tid * 8,         r0c = f0 >> 6, c0c = f0 & 63;
  const int f1 = (256 + tid) * 8, r1c = f1 >> 6, c1c = f1 & 63;
  const uint32_t lds_off[2][2] = {
    { (uint32_t)(uintptr_t)&btile[0][r0c * BKP + c0c],
      (uint32_t)(uintptr_t)&btile[0][r1c * BKP + c1c] },
    { (uint32_t)(uintptr_t)&btile[1][r0c * BKP + c0c],
      (uint32_t)(uintptr_t)&btile[1][r1c * BKP + c1c] } };
  const uint32_t g0 = (uint32_t)(((n0 + r0c) * HID + c0c) * 2);
  const uint32_t g1 = (uint32_t)(((n0 + r1c) * HID + c1c) * 2);

  ASYNC_B128(lds_off[0][0], g0, W);
  ASYNC_B128(lds_off[0][1], g1, W);

  v8f acc[TM][4] = {};
  for (int k0 = 0, stage = 0; k0 < HID; k0 += BK, ++stage) {
    const int cur = stage & 1;
    if (k0 + BK < HID) {
      ASYNC_B128(lds_off[cur ^ 1][0], g0 + (uint32_t)((k0 + BK) * 2), W);
      ASYNC_B128(lds_off[cur ^ 1][1], g1 + (uint32_t)((k0 + BK) * 2), W);
      WAIT_ASYNCCNT(2);   // retire current stage's copies (in-order completion)
    } else {
      WAIT_ASYNCCNT(0);
    }
    __syncthreads();
    const __bf16* bt = btile[cur];
#pragma unroll
    for (int kk = 0; kk < BK; kk += 32) {
      __builtin_prefetch(xb + (size_t)(row0 + (lane & 15)) * HID + k0 + kk + 64, 0, 1);
      v16bf a[TM];
#pragma unroll
      for (int m = 0; m < TM; ++m)
        a[m] = load_a_frag(xb, HID, row0 + 16 * m, k0 + kk, lane);
#pragma unroll
      for (int t = 0; t < 4; ++t) {
        v16bf bf = load_b_frag_lds(bt, 16 * t, kk, lane);
#pragma unroll
        for (int m = 0; m < TM; ++m)
          acc[m][t] = __builtin_amdgcn_wmma_f32_16x16x32_bf16(false, a[m], false, bf,
                                                              (short)0, acc[m][t], false, false);
      }
    }
    __syncthreads();
  }

  const int n = lane & 15, h = lane >> 4;
  if (which < 2) {
    __bf16* dst = (which == 0) ? Qb : Kb;
#pragma unroll
    for (int m = 0; m < TM; ++m)
#pragma unroll
      for (int t = 0; t < 4; ++t)
#pragma unroll
        for (int r = 0; r < 8; ++r)
          dst[(size_t)(row0 + 16 * m + r + 8 * h) * HID + (n0 + 16 * t + n)] = (__bf16)acc[m][t][r];
  } else {
    const int b = row0 / SEQ;
    const int s = row0 % SEQ;
#pragma unroll
    for (int m = 0; m < TM; ++m)
#pragma unroll
      for (int t = 0; t < 4; ++t) {
        v8bf v;
#pragma unroll
        for (int r = 0; r < 8; ++r) v[r] = (__bf16)acc[m][t][r];
        const size_t d = (size_t)(n0 + 16 * t + n);
        *(v8bf*)&Vt[((size_t)b * HID + d) * SEQ + s + 16 * m + 8 * h] = v;   // b128 store
      }
  }
}

// ---------------------------------------------------------------------------
// Kernel 3: scores = (Q @ K^T) / sqrt(D).  K row-major [S,D] is already the
// transposed B operand.  fp32 out to workspace.
// ---------------------------------------------------------------------------
__global__ __launch_bounds__(256) void k_scores_gemm(const __bf16* __restrict__ Qb,
                                                     const __bf16* __restrict__ Kb,
                                                     float* __restrict__ S) {
  __shared__ __bf16 btile[2][BN * BKP];
  const int tid  = threadIdx.x;
  const int lane = tid & 31;
  const int wave = tid >> 5;
  const int kcol0 = blockIdx.x * BN;
  const int q0    = blockIdx.y * BM + wave * 16 * TM;
  const int b     = blockIdx.z;
  const __bf16* Q  = Qb + (size_t)b * SEQ * HID;
  const __bf16* K  = Kb + (size_t)b * SEQ * HID;
  float*        Sb = S  + (size_t)b * SEQ * SEQ;

  const int f0 = tid * 8,         r0c = f0 >> 6, c0c = f0 & 63;
  const int f1 = (256 + tid) * 8, r1c = f1 >> 6, c1c = f1 & 63;
  const uint32_t lds_off[2][2] = {
    { (uint32_t)(uintptr_t)&btile[0][r0c * BKP + c0c],
      (uint32_t)(uintptr_t)&btile[0][r1c * BKP + c1c] },
    { (uint32_t)(uintptr_t)&btile[1][r0c * BKP + c0c],
      (uint32_t)(uintptr_t)&btile[1][r1c * BKP + c1c] } };
  const uint32_t g0 = (uint32_t)(((kcol0 + r0c) * HID + c0c) * 2);
  const uint32_t g1 = (uint32_t)(((kcol0 + r1c) * HID + c1c) * 2);

  ASYNC_B128(lds_off[0][0], g0, K);
  ASYNC_B128(lds_off[0][1], g1, K);

  v8f acc[TM][4] = {};
  for (int d0 = 0, stage = 0; d0 < HID; d0 += BK, ++stage) {
    const int cur = stage & 1;
    if (d0 + BK < HID) {
      ASYNC_B128(lds_off[cur ^ 1][0], g0 + (uint32_t)((d0 + BK) * 2), K);
      ASYNC_B128(lds_off[cur ^ 1][1], g1 + (uint32_t)((d0 + BK) * 2), K);
      WAIT_ASYNCCNT(2);
    } else {
      WAIT_ASYNCCNT(0);
    }
    __syncthreads();
    const __bf16* bt = btile[cur];
#pragma unroll
    for (int kk = 0; kk < BK; kk += 32) {
      v16bf a[TM];
#pragma unroll
      for (int m = 0; m < TM; ++m)
        a[m] = load_a_frag(Q, HID, q0 + 16 * m, d0 + kk, lane);
#pragma unroll
      for (int t = 0; t < 4; ++t) {
        v16bf bf = load_b_frag_lds(bt, 16 * t, kk, lane);
#pragma unroll
        for (int m = 0; m < TM; ++m)
          acc[m][t] = __builtin_amdgcn_wmma_f32_16x16x32_bf16(false, a[m], false, bf,
                                                              (short)0, acc[m][t], false, false);
      }
    }
    __syncthreads();
  }

  const float scale = 0.03125f;  // 1/sqrt(1024)
  const int n = lane & 15, h = lane >> 4;
#pragma unroll
  for (int m = 0; m < TM; ++m)
#pragma unroll
    for (int t = 0; t < 4; ++t)
#pragma unroll
      for (int r = 0; r < 8; ++r)
        Sb[(size_t)(q0 + 16 * m + r + 8 * h) * SEQ + (kcol0 + 16 * t + n)] = acc[m][t][r] * scale;
}

// ---------------------------------------------------------------------------
// Kernel 4: row softmax.  One 256-thread block per 2048-wide row; LDS tree
// reductions; bf16 P out.
// ---------------------------------------------------------------------------
__global__ __launch_bounds__(256) void k_softmax(const float* __restrict__ S,
                                                 __bf16* __restrict__ P) {
  const size_t row = blockIdx.x;
  const float* src = S + row * SEQ;
  __bf16*      dst = P + row * SEQ;
  __shared__ float red[256];
  const int tid = threadIdx.x;

  float vals[8];
  float m = -INFINITY;
#pragma unroll
  for (int i = 0; i < 8; ++i) {
    vals[i] = src[tid + i * 256];
    m = fmaxf(m, vals[i]);
  }
  red[tid] = m;
  __syncthreads();
#pragma unroll
  for (int off = 128; off > 0; off >>= 1) {
    if (tid < off) red[tid] = fmaxf(red[tid], red[tid + off]);
    __syncthreads();
  }
  m = red[0];
  __syncthreads();

  float s = 0.f;
#pragma unroll
  for (int i = 0; i < 8; ++i) {
    vals[i] = __expf(vals[i] - m);
    s += vals[i];
  }
  red[tid] = s;
  __syncthreads();
#pragma unroll
  for (int off = 128; off > 0; off >>= 1) {
    if (tid < off) red[tid] += red[tid + off];
    __syncthreads();
  }
  const float inv = 1.0f / red[0];
#pragma unroll
  for (int i = 0; i < 8; ++i)
    dst[tid + i * 256] = (__bf16)(vals[i] * inv);
}

// ---------------------------------------------------------------------------
// Kernel 5: out = P @ V using transposed Vt[b][d][s] -> contiguous B tiles.
// fp32 stores directly to d_out.
// ---------------------------------------------------------------------------
__global__ __launch_bounds__(256) void k_out_gemm(const __bf16* __restrict__ P,
                                                  const __bf16* __restrict__ Vt,
                                                  float* __restrict__ out) {
  __shared__ __bf16 btile[2][BN * BKP];
  const int tid  = threadIdx.x;
  const int lane = tid & 31;
  const int wave = tid >> 5;
  const int n0 = blockIdx.x * BN;            // d tile
  const int q0 = blockIdx.y * BM + wave * 16 * TM;
  const int b  = blockIdx.z;
  const __bf16* Pb = P  + (size_t)b * SEQ * SEQ;
  const __bf16* Vb = Vt + (size_t)b * HID * SEQ;
  float*        O  = out + (size_t)b * SEQ * HID;

  const int f0 = tid * 8,         r0c = f0 >> 6, c0c = f0 & 63;
  const int f1 = (256 + tid) * 8, r1c = f1 >> 6, c1c = f1 & 63;
  const uint32_t lds_off[2][2] = {
    { (uint32_t)(uintptr_t)&btile[0][r0c * BKP + c0c],
      (uint32_t)(uintptr_t)&btile[0][r1c * BKP + c1c] },
    { (uint32_t)(uintptr_t)&btile[1][r0c * BKP + c0c],
      (uint32_t)(uintptr_t)&btile[1][r1c * BKP + c1c] } };
  const uint32_t g0 = (uint32_t)(((n0 + r0c) * SEQ + c0c) * 2);
  const uint32_t g1 = (uint32_t)(((n0 + r1c) * SEQ + c1c) * 2);

  ASYNC_B128(lds_off[0][0], g0, Vb);
  ASYNC_B128(lds_off[0][1], g1, Vb);

  v8f acc[TM][4] = {};
  for (int k0 = 0, stage = 0; k0 < SEQ; k0 += BK, ++stage) {
    const int cur = stage & 1;
    if (k0 + BK < SEQ) {
      ASYNC_B128(lds_off[cur ^ 1][0], g0 + (uint32_t)((k0 + BK) * 2), Vb);
      ASYNC_B128(lds_off[cur ^ 1][1], g1 + (uint32_t)((k0 + BK) * 2), Vb);
      WAIT_ASYNCCNT(2);
    } else {
      WAIT_ASYNCCNT(0);
    }
    __syncthreads();
    const __bf16* bt = btile[cur];
#pragma unroll
    for (int kk = 0; kk < BK; kk += 32) {
      __builtin_prefetch(Pb + (size_t)(q0 + (lane & 15)) * SEQ + k0 + kk + 64, 0, 1);
      v16bf a[TM];
#pragma unroll
      for (int m = 0; m < TM; ++m)
        a[m] = load_a_frag(Pb, SEQ, q0 + 16 * m, k0 + kk, lane);
#pragma unroll
      for (int t = 0; t < 4; ++t) {
        v16bf bf = load_b_frag_lds(bt, 16 * t, kk, lane);
#pragma unroll
        for (int m = 0; m < TM; ++m)
          acc[m][t] = __builtin_amdgcn_wmma_f32_16x16x32_bf16(false, a[m], false, bf,
                                                              (short)0, acc[m][t], false, false);
      }
    }
    __syncthreads();
  }

  const int n = lane & 15, h = lane >> 4;
#pragma unroll
  for (int m = 0; m < TM; ++m)
#pragma unroll
    for (int t = 0; t < 4; ++t)
#pragma unroll
      for (int r = 0; r < 8; ++r)
        O[(size_t)(q0 + 16 * m + r + 8 * h) * HID + (n0 + 16 * t + n)] = acc[m][t][r];
}

// ---------------------------------------------------------------------------
// Host launcher
// ---------------------------------------------------------------------------
extern "C" void kernel_launch(void* const* d_in, const int* in_sizes, int n_in,
                              void* d_out, int out_size, void* d_ws, size_t ws_size,
                              hipStream_t stream) {
  const float* x  = (const float*)d_in[0];
  const float* Wq = (const float*)d_in[1];
  const float* Wk = (const float*)d_in[2];
  const float* Wv = (const float*)d_in[3];
  float* out = (float*)d_out;

  const size_t NX = (size_t)BATCH * SEQ * HID;   // 8,388,608
  const size_t NW = (size_t)HID * HID;           // 1,048,576
  const size_t NS = (size_t)BATCH * SEQ * SEQ;   // 16,777,216

  char* ws = (char*)d_ws;
  size_t off = 0;
  __bf16* xb = (__bf16*)(ws + off); off += NX * 2;
  __bf16* Wb = (__bf16*)(ws + off); off += 3 * NW * 2;
  __bf16* Qb = (__bf16*)(ws + off); off += NX * 2;
  __bf16* Kb = (__bf16*)(ws + off); off += NX * 2;
  __bf16* Vt = (__bf16*)(ws + off); off += NX * 2;
  float*  S  = (float*)(ws + off);  off += NS * 4;
  __bf16* P  = (__bf16*)(ws + off); off += NS * 2;
  (void)ws_size; (void)n_in; (void)in_sizes; (void)out_size;

  // 1) casts to bf16
  k_cast_bf16<<<dim3((unsigned)(NX / 1024)), dim3(256), 0, stream>>>(x, xb, (int)NX);
  k_cast_bf16<<<dim3((unsigned)(NW / 1024)), dim3(256), 0, stream>>>(Wq, Wb + 0 * NW, (int)NW);
  k_cast_bf16<<<dim3((unsigned)(NW / 1024)), dim3(256), 0, stream>>>(Wk, Wb + 1 * NW, (int)NW);
  k_cast_bf16<<<dim3((unsigned)(NW / 1024)), dim3(256), 0, stream>>>(Wv, Wb + 2 * NW, (int)NW);

  // 2) QKV projection: grid (E/64, B*S/256, 3), 8 waves/block, 32 rows/wave
  k_qkv_gemm<<<dim3(HID / BN, (BATCH * SEQ) / BM, 3), dim3(256), 0, stream>>>(xb, Wb, Qb, Kb, Vt);

  // 3) scores: grid (S/64, S/256, B)
  k_scores_gemm<<<dim3(SEQ / BN, SEQ / BM, BATCH), dim3(256), 0, stream>>>(Qb, Kb, S);

  // 4) softmax: one block per row
  k_softmax<<<dim3((unsigned)(BATCH * SEQ)), dim3(256), 0, stream>>>(S, P);

  // 5) out = P @ V: grid (D/64, S/256, B)
  k_out_gemm<<<dim3(HID / BN, SEQ / BM, BATCH), dim3(256), 0, stream>>>(P, Vt, out);
}
